// MaskedCouplingRQS_34583076667868
// MI455X (gfx1250) — matrix operand order repaired and checked by reading, not compiled, not run on hardware
//
#include <hip/hip_runtime.h>
#include <hip/hip_bf16.h>
#include <math.h>

// ---- types for WMMA operands ----
typedef _Float16     v16h  __attribute__((ext_vector_type(16)));
typedef float        v8f   __attribute__((ext_vector_type(8)));
typedef unsigned int u32x4 __attribute__((ext_vector_type(4)));

union HV { v16h h; u32x4 u[2]; };

#define INPUT_DIM 64
#define MASK_DIM  32
#define TRANS_DIM 32
#define HIDDEN    128
#define NBINS     8
#define SPLINE_P  25          // 3*8+1
#define ROWS      16          // rows per block
#define RMIN      (-5.0f)
#define RMAX      (5.0f)
#define MIN_BIN   1e-4f
#define MIN_SLOPE 1e-4f

// ---------------------------------------------------------------------------
// f32 -> f16 weight conversion (weights then live in L2, reused by all blocks)
// ---------------------------------------------------------------------------
__global__ void cvt_f32_f16(const float* __restrict__ src, _Float16* __restrict__ dst, int n) {
    int i = blockIdx.x * blockDim.x + threadIdx.x;
    if (i < n) dst[i] = (_Float16)src[i];
}

// ---------------------------------------------------------------------------
// One 16x16 output tile of  D = A(16xK, LDS f16) * W(colsxK, global f16)^T
// A fragment layout per CDNA5 ISA (16-bit A 16x32), B fragment = row-major W
// rows (lane n holds W[n0+n, k0+8h..+7] and W[n0+n, k0+8h+16..+23], same k order as A).
// ---------------------------------------------------------------------------
__device__ __forceinline__ v8f wmma_acc_tile(const _Float16* __restrict__ A, int astride,
                                             const _Float16* __restrict__ B, int K, int lane) {
    v8f acc = {0.f,0.f,0.f,0.f,0.f,0.f,0.f,0.f};
    const int m  = lane & 15;
    const int hi = lane >> 4;
    for (int k0 = 0; k0 < K; k0 += 32) {
        HV a, b;
        const _Float16* ap = A + m * astride + k0 + hi * 8;
        a.u[0] = *(const u32x4*)(ap);          // K = k0+hi*8   .. +7
        a.u[1] = *(const u32x4*)(ap + 16);     // K = k0+hi*8+16.. +23
        const _Float16* bp = B + (size_t)m * K + k0 + hi * 8;
        b.u[0] = *(const u32x4*)(bp);          // K = k0+hi*8   .. +7
        b.u[1] = *(const u32x4*)(bp + 16);     // K = k0+hi*8+16.. +23
        acc = __builtin_amdgcn_wmma_f32_16x16x32_f16(false, a.h, false, b.h,
                                                     (short)0, acc, false, false);
    }
    return acc;
}

// ---------------------------------------------------------------------------
// Fully fused: MLP (WMMA) + RQS spline, 16 rows per block, 8 waves.
// ---------------------------------------------------------------------------
__global__ __launch_bounds__(256)
void rqs_fused(const float* __restrict__ x,
               const _Float16* __restrict__ hW1, const float* __restrict__ b1,
               const _Float16* __restrict__ hW2, const float* __restrict__ b2,
               const _Float16* __restrict__ hW3, const float* __restrict__ b3,
               const _Float16* __restrict__ hW4, const float* __restrict__ b4,
               float* __restrict__ out, int N) {
    __shared__ __align__(16) _Float16 s_x [ROWS * MASK_DIM];
    __shared__ __align__(16) _Float16 s_hA[ROWS * HIDDEN];
    __shared__ __align__(16) _Float16 s_hB[ROWS * HIDDEN];
    __shared__ __align__(16) float    s_p [ROWS * TRANS_DIM * SPLINE_P]; // 16 x 800
    __shared__              float    s_ld[ROWS];

    const int tid  = threadIdx.x;
    const int lane = tid & 31;
    const int w    = tid >> 5;
    const int rb   = blockIdx.x * ROWS;
    const int n    = lane & 15;
    const int mb   = (lane >> 4) * 8;

    if (tid < ROWS) s_ld[tid] = 0.f;

    // stage x_masked in f16, pass it through to output unchanged
    for (int p = tid; p < ROWS * MASK_DIM; p += 256) {
        int r = p >> 5, c = p & 31;
        float v = x[(size_t)(rb + r) * INPUT_DIM + c];
        s_x[r * MASK_DIM + c] = (_Float16)v;
        out[(size_t)(rb + r) * INPUT_DIM + c] = v;
    }
    __syncthreads();

    // ---- layer 1: [16x32] @ W1[128x32]^T -> relu -> s_hA [16x128]
    {
        int n0 = w * 16;
        v8f acc = wmma_acc_tile(s_x, MASK_DIM, hW1 + (size_t)n0 * MASK_DIM, MASK_DIM, lane);
        float bn = b1[n0 + n];
        #pragma unroll
        for (int r = 0; r < 8; ++r)
            s_hA[(mb + r) * HIDDEN + n0 + n] = (_Float16)fmaxf(acc[r] + bn, 0.f);
    }
    __syncthreads();

    // ---- layer 2: s_hA @ W2^T -> relu -> s_hB
    {
        int n0 = w * 16;
        v8f acc = wmma_acc_tile(s_hA, HIDDEN, hW2 + (size_t)n0 * HIDDEN, HIDDEN, lane);
        float bn = b2[n0 + n];
        #pragma unroll
        for (int r = 0; r < 8; ++r)
            s_hB[(mb + r) * HIDDEN + n0 + n] = (_Float16)fmaxf(acc[r] + bn, 0.f);
    }
    __syncthreads();

    // ---- layer 3: s_hB @ W3^T -> relu -> s_hA
    {
        int n0 = w * 16;
        v8f acc = wmma_acc_tile(s_hB, HIDDEN, hW3 + (size_t)n0 * HIDDEN, HIDDEN, lane);
        float bn = b3[n0 + n];
        #pragma unroll
        for (int r = 0; r < 8; ++r)
            s_hA[(mb + r) * HIDDEN + n0 + n] = (_Float16)fmaxf(acc[r] + bn, 0.f);
    }
    __syncthreads();

    // ---- layer 4: s_hA @ W4[800x128]^T + b4 -> s_p (f32, 16 x 800), 50 tiles
    for (int t = w; t < 50; t += 8) {
        int n0 = t * 16;
        v8f acc = wmma_acc_tile(s_hA, HIDDEN, hW4 + (size_t)n0 * HIDDEN, HIDDEN, lane);
        float bn = b4[n0 + n];
        #pragma unroll
        for (int r = 0; r < 8; ++r)
            s_p[(mb + r) * (TRANS_DIM * SPLINE_P) + n0 + n] = acc[r] + bn;
    }
    __syncthreads();

    // ---- rational-quadratic spline: 512 (row, feature) pairs
    for (int p = tid; p < ROWS * TRANS_DIM; p += 256) {
        int r = p >> 5, t = p & 31;
        const float* raw = &s_p[r * (TRANS_DIM * SPLINE_P) + t * SPLINE_P];
        float rv[SPLINE_P];
        #pragma unroll
        for (int j = 0; j < SPLINE_P; ++j) rv[j] = raw[j];

        // softmax widths / heights (stable)
        float mw = rv[0], mh = rv[8];
        #pragma unroll
        for (int j = 1; j < 8; ++j) { mw = fmaxf(mw, rv[j]); mh = fmaxf(mh, rv[8 + j]); }
        float ew[8], eh[8], sw = 0.f, sh = 0.f;
        #pragma unroll
        for (int j = 0; j < 8; ++j) {
            ew[j] = __expf(rv[j] - mw);     sw += ew[j];
            eh[j] = __expf(rv[8 + j] - mh); sh += eh[j];
        }
        const float scale = (RMAX - RMIN) - NBINS * MIN_BIN;
        float wd[8], ht[8];
        #pragma unroll
        for (int j = 0; j < 8; ++j) {
            wd[j] = MIN_BIN + scale * ew[j] / sw;
            ht[j] = MIN_BIN + scale * eh[j] / sh;
        }
        float sl[9];
        #pragma unroll
        for (int j = 0; j < 9; ++j) {
            float z  = rv[16 + j];
            float sp = (z > 20.f) ? z : __logf(1.f + __expf(z));
            sl[j] = MIN_SLOPE + sp;
        }
        float cw[9], ch[9]; cw[0] = 0.f; ch[0] = 0.f;
        #pragma unroll
        for (int j = 0; j < 8; ++j) { cw[j + 1] = cw[j] + wd[j]; ch[j + 1] = ch[j] + ht[j]; }

        float xv = x[(size_t)(rb + r) * INPUT_DIM + MASK_DIM + t];
        bool inside = (xv >= RMIN) && (xv <= RMAX);
        float xc = fminf(fmaxf(xv, RMIN), RMAX);

        int idx = 0;
        #pragma unroll
        for (int j = 1; j < 8; ++j) if (xc >= RMIN + cw[j]) idx = j;

        // gather bin quantities via cndmask select chains (no scratch)
        float xk = cw[0], yk = ch[0], wk = wd[0], hk = ht[0], dk = sl[0], dk1 = sl[1];
        #pragma unroll
        for (int j = 1; j < 8; ++j) {
            bool msel = (idx == j);
            xk  = msel ? cw[j]     : xk;
            yk  = msel ? ch[j]     : yk;
            wk  = msel ? wd[j]     : wk;
            hk  = msel ? ht[j]     : hk;
            dk  = msel ? sl[j]     : dk;
            dk1 = msel ? sl[j + 1] : dk1;
        }
        xk += RMIN; yk += RMIN;

        float xi  = (xc - xk) / wk;
        float om  = 1.f - xi;
        float sk  = hk / wk;
        float den = sk + (dk1 + dk - 2.f * sk) * xi * om;
        float y_in  = yk + hk * (sk * xi * xi + dk * xi * om) / den;
        float num   = sk * sk * (dk1 * xi * xi + 2.f * sk * xi * om + dk * om * om);
        float ld_in = __logf(num) - 2.f * __logf(den);

        float yv = inside ? y_in  : xv;
        float ld = inside ? ld_in : 0.f;
        out[(size_t)(rb + r) * INPUT_DIM + MASK_DIM + t] = yv;
        atomicAdd(&s_ld[r], ld);   // ds_add_f32
    }
    __syncthreads();
    if (tid < ROWS) out[(size_t)N * INPUT_DIM + rb + tid] = s_ld[tid];
}

// ---------------------------------------------------------------------------
extern "C" void kernel_launch(void* const* d_in, const int* in_sizes, int n_in,
                              void* d_out, int out_size, void* d_ws, size_t ws_size,
                              hipStream_t stream) {
    const float* x  = (const float*)d_in[0];
    const float* W1 = (const float*)d_in[1];
    const float* b1 = (const float*)d_in[2];
    const float* W2 = (const float*)d_in[3];
    const float* b2 = (const float*)d_in[4];
    const float* W3 = (const float*)d_in[5];
    const float* b3 = (const float*)d_in[6];
    const float* W4 = (const float*)d_in[7];
    const float* b4 = (const float*)d_in[8];
    float* out = (float*)d_out;

    const int N = in_sizes[0] / INPUT_DIM;   // 262144 rows

    // f16 weight staging in workspace
    _Float16* ws  = (_Float16*)d_ws;
    _Float16* hW1 = ws;                               // 128*32   = 4096
    _Float16* hW2 = hW1 + HIDDEN * MASK_DIM;          // 128*128  = 16384
    _Float16* hW3 = hW2 + HIDDEN * HIDDEN;            // 128*128  = 16384
    _Float16* hW4 = hW3 + HIDDEN * HIDDEN;            // 800*128  = 102400

    const int n1 = HIDDEN * MASK_DIM;
    const int n2 = HIDDEN * HIDDEN;
    const int n4 = TRANS_DIM * SPLINE_P * HIDDEN;
    cvt_f32_f16<<<(n1 + 255) / 256, 256, 0, stream>>>(W1, hW1, n1);
    cvt_f32_f16<<<(n2 + 255) / 256, 256, 0, stream>>>(W2, hW2, n2);
    cvt_f32_f16<<<(n2 + 255) / 256, 256, 0, stream>>>(W3, hW3, n2);
    cvt_f32_f16<<<(n4 + 255) / 256, 256, 0, stream>>>(W4, hW4, n4);

    rqs_fused<<<N / ROWS, 256, 0, stream>>>(x, hW1, b1, hW2, b2, hW3, b3, hW4, b4, out, N);
}
